// GAUBlock_41326175322889
// MI455X (gfx1250) — matrix operand order repaired and checked by reading, not compiled, not run on hardware
//
#include <hip/hip_runtime.h>
#include <hip/hip_bf16.h>

#define D_DIM 1024
#define E_DIM 2048
#define S_DIM 128
#define F_DIM (2 * E_DIM + S_DIM)   // 4224
#define L_DIM 2048
#define B_DIM 4
#define M_ROWS (B_DIM * L_DIM)      // 8192

typedef __attribute__((ext_vector_type(16))) __bf16 v16bf;
typedef __attribute__((ext_vector_type(8)))  float  v8f;
typedef __attribute__((ext_vector_type(4)))  unsigned int v4u;
typedef __attribute__((ext_vector_type(8)))  int    v8i;
typedef __attribute__((ext_vector_type(4)))  int    v4i;

__device__ __forceinline__ unsigned short f2bf(float x) {
    unsigned u = __float_as_uint(x);
    u += 0x7fffu + ((u >> 16) & 1u);      // round-to-nearest-even
    return (unsigned short)(u >> 16);
}
__device__ __forceinline__ float bf2f(unsigned short h) {
    return __uint_as_float(((unsigned)h) << 16);
}

// ---------------------------------------------------------------------------
// Tensor Data Mover: DMA a 2D tile (rows x kcols bf16, row stride ld elems)
// from global memory into LDS (packed rows, kcols*2 bytes each).
// D# layout per CDNA5 ISA.
// ---------------------------------------------------------------------------
#if __has_builtin(__builtin_amdgcn_tensor_load_to_lds)
#define USE_TDM 1
__device__ __forceinline__ void tdm_load_2d(unsigned lds_addr, const void* gaddr,
                                            int ld_elems, int rows, int kcols) {
    const unsigned long long ga = (unsigned long long)gaddr;
    v4u g0;
    g0.x = 1u;                                            // count=1, user mode
    g0.y = lds_addr;                                      // LDS byte address
    g0.z = (unsigned)ga;                                  // global_addr[31:0]
    g0.w = (unsigned)((ga >> 32) & 0x01FFFFFFu) | (2u << 30);  // addr[56:32], type=2
    v8i g1;
    g1[0] = 0x00010000;               // wg_mask=0, data_size=1 (2 bytes)
    g1[1] = 0;                        // tensor_dim0 = 0x40000000 -> [15:0]<<16 = 0
    g1[2] = 0x4000;                   // td0[31:16] | td1[15:0]<<16 (td1=0x40000000)
    g1[3] = 0x4000 | (kcols << 16);   // td1[31:16] | tile_dim0
    g1[4] = rows;                     // tile_dim1 | tile_dim2=0
    g1[5] = ld_elems;                 // tensor_dim0_stride[31:0]
    g1[6] = 0;                        // stride0[47:32] | stride1[15:0]
    g1[7] = 0;                        // stride1[47:16]
    const v4i gz = {0, 0, 0, 0};
#if __clang_major__ >= 23
    const v8i gz8 = {0, 0, 0, 0, 0, 0, 0, 0};
    __builtin_amdgcn_tensor_load_to_lds(g0, g1, gz, gz, gz8, 0);
#else
    __builtin_amdgcn_tensor_load_to_lds(g0, g1, gz, gz, 0);
#endif
}
__device__ __forceinline__ void wait_tdm() {
    __builtin_amdgcn_s_wait_tensorcnt(0);
}
#else
#define USE_TDM 0
#endif

// Fallback: per-lane CDNA5 async copy global -> LDS (ASYNCcnt).
__device__ __forceinline__ void async_copy_b128(unsigned lds_off, const void* gaddr) {
    asm volatile("global_load_async_to_lds_b128 %0, %1, off"
                 :: "v"(lds_off), "v"(gaddr) : "memory");
}
__device__ __forceinline__ void wait_async() {
    asm volatile("s_wait_asynccnt 0x0" ::: "memory");
}

__device__ __forceinline__ v8f wmma_bf16(v16bf a, v16bf b, v8f c) {
    return __builtin_amdgcn_wmma_f32_16x16x32_bf16(false, a, false, b,
                                                   (short)0, c, false, false);
}

// ---------------------------------------------------------------------------
// A fragment: 16x32 bf16 row-major tile (lda in elements).
// lanes 0-15 -> M=lane, K={0..7,16..23}; lanes 16-31 -> M=lane-16,
// K={8..15,24..31}; each VGPR packs 2 consecutive K.
// ---------------------------------------------------------------------------
__device__ __forceinline__ v16bf load_a_frag(const unsigned short* A, int lda) {
    const int lane = threadIdx.x & 31;
    const int m    = lane & 15;
    const int kb   = (lane >> 4) << 3;          // 0 or 8
    union { v16bf v; unsigned u[8]; } f;
    const unsigned short* p = A + (size_t)m * lda + kb;
#pragma unroll
    for (int g = 0; g < 4; ++g) f.u[g]     = *(const unsigned*)(p + 2 * g);
#pragma unroll
    for (int g = 0; g < 4; ++g) f.u[4 + g] = *(const unsigned*)(p + 16 + 2 * g);
    return f.v;
}

// B fragment: 32x16 from Bt[n][k] (K contiguous). lanes 0-15 -> N=lane,
// K={0..7,16..23}; lanes 16-31 -> N=lane-16, K={8..15,24..31}.
__device__ __forceinline__ v16bf load_b_frag(const unsigned short* Bt, int ldb) {
    const int lane = threadIdx.x & 31;
    const int n    = lane & 15;
    const int kb   = (lane >> 4) << 3;          // 0 or 8
    union { v16bf v; unsigned u[8]; } f;
    const unsigned short* p = Bt + (size_t)n * ldb + kb;
#pragma unroll
    for (int g = 0; g < 4; ++g) f.u[g]     = *(const unsigned*)(p + 2 * g);
#pragma unroll
    for (int g = 0; g < 4; ++g) f.u[4 + g] = *(const unsigned*)(p + 16 + 2 * g);
    return f.v;
}

// ---------------------------------------------------------------------------
// Block GEMM: 128(M) x 128(N) tile, K-chunks of 64, TDM double-buffered LDS.
// Ablk = A + m0*lda (row-major, K contiguous); Bblk = Bt + n0*ldb (Bt[n][k]).
// 256 threads = 8 waves; wave (w&3 -> 32-row sub, w>>2 -> 64-col sub).
// ---------------------------------------------------------------------------
__device__ __forceinline__ void block_gemm_128(const unsigned short* __restrict__ Ablk, int lda,
                                               const unsigned short* __restrict__ Bblk, int ldb,
                                               int K, v8f acc[2][4]) {
    __shared__ unsigned short As[2][128][64];
    __shared__ unsigned short Bs[2][128][64];
    const int tid = threadIdx.x;
    const int w   = tid >> 5;
    const int mi  = (w & 3) * 32;
    const int ni  = (w >> 2) * 64;

    const unsigned asb = (unsigned)(size_t)(void*)&As[0][0][0];
    const unsigned bsb = (unsigned)(size_t)(void*)&Bs[0][0][0];
    const int nch = K >> 6;

#if USE_TDM
    // One wave drives the Tensor Data Mover; panels arrive via TENSORcnt.
    const bool issuer = (w == 0);
    auto issue = [&](int chunk, int bb) {
        tdm_load_2d(asb + (unsigned)bb * (128 * 64 * 2), Ablk + chunk * 64, lda, 128, 64);
        tdm_load_2d(bsb + (unsigned)bb * (128 * 64 * 2), Bblk + chunk * 64, ldb, 128, 64);
    };
    if (issuer) issue(0, 0);
    for (int c = 0; c < nch; ++c) {
        const int bb = c & 1;
        if (issuer) wait_tdm();   // chunk c landed in LDS
        __syncthreads();          // publish; chunk c-1 consumers are done
        if (issuer && c + 1 < nch) issue(c + 1, bb ^ 1);
#pragma unroll
        for (int ks = 0; ks < 2; ++ks) {
            v16bf a0 = load_a_frag(&As[bb][mi][ks * 32], 64);
            v16bf a1 = load_a_frag(&As[bb][mi + 16][ks * 32], 64);
#pragma unroll
            for (int t = 0; t < 4; ++t) {
                v16bf b = load_b_frag(&Bs[bb][ni + t * 16][ks * 32], 64);
                acc[0][t] = wmma_bf16(a0, b, acc[0][t]);
                acc[1][t] = wmma_bf16(a1, b, acc[1][t]);
            }
        }
        __syncthreads();          // done with buf bb before TDM refills it
    }
#else
    // Fallback: cooperative per-lane async copies (ASYNCcnt).
    const int cr = tid >> 3;            // copy row 0..31
    const int cs = (tid & 7) * 8;       // half-offset within row (16B segs)
    auto issue = [&](int chunk, int bb) {
        const unsigned short* Ak = Ablk + chunk * 64;
        const unsigned short* Bk = Bblk + chunk * 64;
#pragma unroll
        for (int p = 0; p < 4; ++p) {
            const int r = p * 32 + cr;
            async_copy_b128(asb + (((unsigned)bb * 128 + r) * 64 + cs) * 2,
                            Ak + (size_t)r * lda + cs);
            async_copy_b128(bsb + (((unsigned)bb * 128 + r) * 64 + cs) * 2,
                            Bk + (size_t)r * ldb + cs);
        }
    };
    issue(0, 0);
    for (int c = 0; c < nch; ++c) {
        const int bb = c & 1;
        wait_async();
        __syncthreads();
        if (c + 1 < nch) issue(c + 1, bb ^ 1);
#pragma unroll
        for (int ks = 0; ks < 2; ++ks) {
            v16bf a0 = load_a_frag(&As[bb][mi][ks * 32], 64);
            v16bf a1 = load_a_frag(&As[bb][mi + 16][ks * 32], 64);
#pragma unroll
            for (int t = 0; t < 4; ++t) {
                v16bf b = load_b_frag(&Bs[bb][ni + t * 16][ks * 32], 64);
                acc[0][t] = wmma_bf16(a0, b, acc[0][t]);
                acc[1][t] = wmma_bf16(a1, b, acc[1][t]);
            }
        }
    }
#endif
}

// ---------------------------------------------------------------------------
// Kernel 1: LayerNorm -> bf16. One 256-thread block per row of 1024.
// ---------------------------------------------------------------------------
__global__ void ln_to_bf16(const float* __restrict__ x,
                           const float* __restrict__ g,
                           const float* __restrict__ b,
                           unsigned short* __restrict__ xn) {
    __shared__ float s1[256], s2[256];
    const int row = blockIdx.x, tid = threadIdx.x;
    const float* xr = x + (size_t)row * D_DIM;
    float v[4], s = 0.f, sq = 0.f;
#pragma unroll
    for (int i = 0; i < 4; ++i) {
        float t = xr[tid + 256 * i];
        v[i] = t; s += t; sq += t * t;
    }
    s1[tid] = s; s2[tid] = sq; __syncthreads();
    for (int off = 128; off > 0; off >>= 1) {
        if (tid < off) { s1[tid] += s1[tid + off]; s2[tid] += s2[tid + off]; }
        __syncthreads();
    }
    const float mu   = s1[0] * (1.f / D_DIM);
    const float var  = s2[0] * (1.f / D_DIM) - mu * mu;
    const float rstd = rsqrtf(var + 1e-5f);
#pragma unroll
    for (int i = 0; i < 4; ++i) {
        int c = tid + 256 * i;
        xn[(size_t)row * D_DIM + c] = f2bf((v[i] - mu) * rstd * g[c] + b[c]);
    }
}

// Transpose + f32->bf16:  Wt[c][r] = bf16(W[r][c]),  W is RxC row-major.
__global__ void transpose_cvt(const float* __restrict__ W,
                              unsigned short* __restrict__ Wt,
                              int R, int C) {
    size_t i = (size_t)blockIdx.x * 256 + threadIdx.x;
    if (i >= (size_t)R * C) return;
    int r = (int)(i / C), c = (int)(i % C);
    Wt[(size_t)c * R + r] = f2bf(W[i]);
}

// ---------------------------------------------------------------------------
// GEMM1: uvz = silu(xn @ W_in + b_in); split into u, vT, q/k (affine).
// M=8192, N=4224, K=1024. grid (33, 64).
// ---------------------------------------------------------------------------
__global__ void gemm_in(const unsigned short* __restrict__ xn,
                        const unsigned short* __restrict__ WtIn,
                        const float* __restrict__ b_in,
                        const float* __restrict__ gq, const float* __restrict__ bq,
                        const float* __restrict__ gk, const float* __restrict__ bk,
                        unsigned short* __restrict__ u,
                        unsigned short* __restrict__ vT,
                        unsigned short* __restrict__ q,
                        unsigned short* __restrict__ kmat) {
    const int m0 = blockIdx.y * 128;
    const int n0 = blockIdx.x * 128;
    v8f acc[2][4] = {};
    block_gemm_128(xn + (size_t)m0 * D_DIM, D_DIM,
                   WtIn + (size_t)n0 * D_DIM, D_DIM, D_DIM, acc);
    const int w = threadIdx.x >> 5, lane = threadIdx.x & 31;
    const int nn = lane & 15, mo = (lane >> 4) << 3;
#pragma unroll
    for (int sub = 0; sub < 2; ++sub) {
#pragma unroll
        for (int t = 0; t < 4; ++t) {
            const int n = n0 + (w >> 2) * 64 + t * 16 + nn;
            const float bias = b_in[n];
#pragma unroll
            for (int r = 0; r < 8; ++r) {
                const int row = m0 + (w & 3) * 32 + sub * 16 + mo + r;
                const float val = acc[sub][t][r] + bias;
                const float s = val / (1.f + __expf(-val));      // SiLU
                if (n < E_DIM) {
                    u[(size_t)row * E_DIM + n] = f2bf(s);
                } else if (n < 2 * E_DIM) {
                    const int e = n - E_DIM, bb = row >> 11, l = row & (L_DIM - 1);
                    vT[((size_t)bb * E_DIM + e) * L_DIM + l] = f2bf(s);
                } else {
                    const int si = n - 2 * E_DIM;
                    q[(size_t)row * S_DIM + si]    = f2bf(s * gq[si] + bq[si]);
                    kmat[(size_t)row * S_DIM + si] = f2bf(s * gk[si] + bk[si]);
                }
            }
        }
    }
}

// ---------------------------------------------------------------------------
// GEMM2: attn = relu((q @ k^T)/L)^2 per batch. M=N=2048, K=128. grid (16,16,4)
// ---------------------------------------------------------------------------
__global__ void gemm_attn(const unsigned short* __restrict__ q,
                          const unsigned short* __restrict__ kmat,
                          unsigned short* __restrict__ attn) {
    const int b  = blockIdx.z;
    const int m0 = blockIdx.y * 128;
    const int n0 = blockIdx.x * 128;
    const unsigned short* qb = q    + (size_t)b * L_DIM * S_DIM;
    const unsigned short* kb = kmat + (size_t)b * L_DIM * S_DIM;
    v8f acc[2][4] = {};
    block_gemm_128(qb + (size_t)m0 * S_DIM, S_DIM,
                   kb + (size_t)n0 * S_DIM, S_DIM, S_DIM, acc);
    unsigned short* ab = attn + (size_t)b * L_DIM * L_DIM;
    const int w = threadIdx.x >> 5, lane = threadIdx.x & 31;
    const int nn = lane & 15, mo = (lane >> 4) << 3;
#pragma unroll
    for (int sub = 0; sub < 2; ++sub) {
#pragma unroll
        for (int t = 0; t < 4; ++t) {
            const int n = n0 + (w >> 2) * 64 + t * 16 + nn;
#pragma unroll
            for (int r = 0; r < 8; ++r) {
                const int row = m0 + (w & 3) * 32 + sub * 16 + mo + r;
                float a = acc[sub][t][r] * (1.f / (float)L_DIM);
                a = fmaxf(a, 0.f);
                ab[(size_t)row * L_DIM + n] = f2bf(a * a);
            }
        }
    }
}

// ---------------------------------------------------------------------------
// GEMM3: out = u * (attn @ v) per batch. M=N=2048, K=2048. grid (16,16,4)
// ---------------------------------------------------------------------------
__global__ void gemm_pv(const unsigned short* __restrict__ attn,
                        const unsigned short* __restrict__ vT,
                        const unsigned short* __restrict__ u,
                        unsigned short* __restrict__ outm) {
    const int b  = blockIdx.z;
    const int m0 = blockIdx.y * 128;
    const int n0 = blockIdx.x * 128;
    const unsigned short* ab = attn + (size_t)b * L_DIM * L_DIM;
    const unsigned short* vb = vT   + (size_t)b * E_DIM * L_DIM;
    v8f acc[2][4] = {};
    block_gemm_128(ab + (size_t)m0 * L_DIM, L_DIM,
                   vb + (size_t)n0 * L_DIM, L_DIM, L_DIM, acc);
    const int w = threadIdx.x >> 5, lane = threadIdx.x & 31;
    const int nn = lane & 15, mo = (lane >> 4) << 3;
#pragma unroll
    for (int sub = 0; sub < 2; ++sub) {
#pragma unroll
        for (int t = 0; t < 4; ++t) {
            const int n = n0 + (w >> 2) * 64 + t * 16 + nn;
#pragma unroll
            for (int r = 0; r < 8; ++r) {
                const size_t grow = (size_t)b * L_DIM + (m0 + (w & 3) * 32 + sub * 16 + mo + r);
                const float uu = bf2f(u[grow * E_DIM + n]);
                outm[grow * E_DIM + n] = f2bf(acc[sub][t][r] * uu);
            }
        }
    }
}

// ---------------------------------------------------------------------------
// GEMM4: y = x + out @ W_out + b_out. M=8192, N=1024, K=2048. grid (8, 64)
// ---------------------------------------------------------------------------
__global__ void gemm_out(const unsigned short* __restrict__ outm,
                         const unsigned short* __restrict__ WtOut,
                         const float* __restrict__ b_out,
                         const float* __restrict__ x,
                         float* __restrict__ y) {
    const int m0 = blockIdx.y * 128;
    const int n0 = blockIdx.x * 128;
    v8f acc[2][4] = {};
    block_gemm_128(outm + (size_t)m0 * E_DIM, E_DIM,
                   WtOut + (size_t)n0 * E_DIM, E_DIM, E_DIM, acc);
    const int w = threadIdx.x >> 5, lane = threadIdx.x & 31;
    const int nn = lane & 15, mo = (lane >> 4) << 3;
#pragma unroll
    for (int sub = 0; sub < 2; ++sub) {
#pragma unroll
        for (int t = 0; t < 4; ++t) {
            const int n = n0 + (w >> 2) * 64 + t * 16 + nn;
            const float bias = b_out[n];
#pragma unroll
            for (int r = 0; r < 8; ++r) {
                const size_t row = (size_t)(m0 + (w & 3) * 32 + sub * 16 + mo + r);
                y[row * D_DIM + n] = acc[sub][t][r] + bias + x[row * D_DIM + n];
            }
        }
    }
}

// ---------------------------------------------------------------------------
extern "C" void kernel_launch(void* const* d_in, const int* in_sizes, int n_in,
                              void* d_out, int out_size, void* d_ws, size_t ws_size,
                              hipStream_t stream) {
    const float* x     = (const float*)d_in[0];
    const float* ln_g  = (const float*)d_in[1];
    const float* ln_b  = (const float*)d_in[2];
    const float* W_in  = (const float*)d_in[3];
    const float* b_in  = (const float*)d_in[4];
    const float* W_out = (const float*)d_in[5];
    const float* b_out = (const float*)d_in[6];
    const float* gq    = (const float*)d_in[7];
    const float* bq    = (const float*)d_in[8];
    const float* gk    = (const float*)d_in[9];
    const float* bk    = (const float*)d_in[10];

    char* ws = (char*)d_ws;
    size_t off = 0;
    auto alloc = [&](size_t elems) {
        unsigned short* p = (unsigned short*)(ws + off);
        off += (elems * sizeof(unsigned short) + 255) & ~(size_t)255;
        return p;
    };
    unsigned short* xn    = alloc((size_t)M_ROWS * D_DIM);
    unsigned short* WtIn  = alloc((size_t)F_DIM * D_DIM);
    unsigned short* WtOut = alloc((size_t)D_DIM * E_DIM);
    unsigned short* u     = alloc((size_t)M_ROWS * E_DIM);
    unsigned short* vT    = alloc((size_t)B_DIM * E_DIM * L_DIM);
    unsigned short* q     = alloc((size_t)M_ROWS * S_DIM);
    unsigned short* kmat  = alloc((size_t)M_ROWS * S_DIM);
    unsigned short* attn  = alloc((size_t)B_DIM * L_DIM * L_DIM);
    unsigned short* outm  = alloc((size_t)M_ROWS * E_DIM);

    ln_to_bf16<<<M_ROWS, 256, 0, stream>>>(x, ln_g, ln_b, xn);
    transpose_cvt<<<((size_t)D_DIM * F_DIM + 255) / 256, 256, 0, stream>>>(W_in, WtIn, D_DIM, F_DIM);
    transpose_cvt<<<((size_t)E_DIM * D_DIM + 255) / 256, 256, 0, stream>>>(W_out, WtOut, E_DIM, D_DIM);

    gemm_in<<<dim3(F_DIM / 128, M_ROWS / 128), 256, 0, stream>>>(
        xn, WtIn, b_in, gq, bq, gk, bk, u, vT, q, kmat);
    gemm_attn<<<dim3(L_DIM / 128, L_DIM / 128, B_DIM), 256, 0, stream>>>(q, kmat, attn);
    gemm_pv<<<dim3(E_DIM / 128, L_DIM / 128, B_DIM), 256, 0, stream>>>(attn, vT, u, outm);
    gemm_out<<<dim3(D_DIM / 128, M_ROWS / 128), 256, 0, stream>>>(outm, WtOut, b_out, x, (float*)d_out);
}
